// _Attention_Geometric_Backflow_GPU_40870908789107
// MI455X (gfx1250) — matrix operand order repaired and checked by reading, not compiled, not run on hardware
//
#include <hip/hip_runtime.h>

typedef _Float16 f16;
typedef __attribute__((ext_vector_type(16))) _Float16 v16h;
typedef __attribute__((ext_vector_type(8)))  _Float16 h8;
typedef __attribute__((ext_vector_type(4)))  _Float16 h4;
typedef __attribute__((ext_vector_type(8)))  float    v8f;

#define SS   256   // sites
#define DD   256   // model dim
#define HH   8     // heads
#define HDIM 32    // head dim
#define NLAY 4
#define FFD  1024
#define BB   64    // batch
#define NROWS (BB*SS)   // 16384

// ---------------------------------------------------------------- WMMA helper
__device__ inline v8f wmma16(v16h a, v16h b, v8f c) {
  return __builtin_amdgcn_wmma_f32_16x16x32_f16(false, a, false, b, (short)0, c,
                                                false, false);
}

// A-fragment (16x32 f16): lanes 0-15 m=lane, k = 2*off + e (contiguous 16 halves)
// with off = (lane<16 ? 0 : 8).  Two contiguous 8-half (16B) loads.
__device__ inline v16h frag_a(const f16* row, int off) {
  h8 lo = *(const h8*)(row + 2 * off);
  h8 hi = *(const h8*)(row + 2 * off + 8);
  v16h r;
#pragma unroll
  for (int i = 0; i < 8; ++i) { r[i] = lo[i]; r[i + 8] = hi[i]; }
  return r;
}

// B-fragment (32x16 f16): lane n=lane%16, k = (lane<16?0:16)+e, contiguous 16 halves.
__device__ inline v16h frag_b(const f16* row) {
  h8 lo = *(const h8*)(row);
  h8 hi = *(const h8*)(row + 8);
  v16h r;
#pragma unroll
  for (int i = 0; i < 8; ++i) { r[i] = lo[i]; r[i + 8] = hi[i]; }
  return r;
}

// ---------------------------------------------------------------- async LDS copy
// GLOBAL_LOAD_ASYNC_TO_LDS_B128 (GV mode): vdst = LDS byte address (low 32 bits
// of the flat shared pointer, per the aperture mapping), vaddr = 64-bit global
// address, saddr = off.  Tracked with ASYNCcnt.
__device__ inline void async_load_b128(const void* gsrc, void* ldsdst) {
  unsigned int l = (unsigned int)(size_t)ldsdst;
  asm volatile("global_load_async_to_lds_b128 %0, %1, off"
               :: "v"(l), "v"(gsrc) : "memory");
}

__device__ inline void wait_async0() {
  asm volatile("s_wait_asynccnt 0x0" ::: "memory");
}

// ---------------------------------------------------------------- small kernels
__global__ void k_f32_to_f16(const float* __restrict__ in, f16* __restrict__ out,
                             int n) {
  int i = blockIdx.x * 256 + threadIdx.x;
  if (i < n) out[i] = (f16)in[i];
}

__global__ void k_embed(const int* __restrict__ x,
                        const float* __restrict__ emb,
                        float* __restrict__ h) {
  int row = blockIdx.x;          // b*S + s
  int s   = row % SS;
  int xi  = x[row];
  h[(long)row * DD + threadIdx.x] = emb[((long)s * 2 + xi) * DD + threadIdx.x];
}

// wave-per-row LayerNorm over D=256, f32 in -> f16 out
__global__ void k_ln(const float* __restrict__ in, const float* __restrict__ g,
                     const float* __restrict__ b, f16* __restrict__ out,
                     int nrows) {
  int wid = threadIdx.x >> 5, lane = threadIdx.x & 31;
  int row = blockIdx.x * 8 + wid;
  if (row >= nrows) return;
  const float4* p = (const float4*)(in + (long)row * DD);
  float4 a0 = p[lane], a1 = p[lane + 32];
  float s = a0.x + a0.y + a0.z + a0.w + a1.x + a1.y + a1.z + a1.w;
#pragma unroll
  for (int m = 16; m >= 1; m >>= 1) s += __shfl_xor(s, m, 32);
  float mu = s * (1.0f / 256.0f);
  float d0 = a0.x - mu, d1 = a0.y - mu, d2 = a0.z - mu, d3 = a0.w - mu;
  float d4 = a1.x - mu, d5 = a1.y - mu, d6 = a1.z - mu, d7 = a1.w - mu;
  float vs = d0*d0 + d1*d1 + d2*d2 + d3*d3 + d4*d4 + d5*d5 + d6*d6 + d7*d7;
#pragma unroll
  for (int m = 16; m >= 1; m >>= 1) vs += __shfl_xor(vs, m, 32);
  float inv = rsqrtf(vs * (1.0f / 256.0f) + 1e-5f);
  const float4* gp = (const float4*)g;
  const float4* bp = (const float4*)b;
  float4 g0 = gp[lane], g1 = gp[lane + 32];
  float4 b0 = bp[lane], b1 = bp[lane + 32];
  f16* o = out + (long)row * DD;
  h4 o0, o1;
  o0[0] = (f16)(d0 * inv * g0.x + b0.x);
  o0[1] = (f16)(d1 * inv * g0.y + b0.y);
  o0[2] = (f16)(d2 * inv * g0.z + b0.z);
  o0[3] = (f16)(d3 * inv * g0.w + b0.w);
  o1[0] = (f16)(d4 * inv * g1.x + b1.x);
  o1[1] = (f16)(d5 * inv * g1.y + b1.y);
  o1[2] = (f16)(d6 * inv * g1.z + b1.z);
  o1[3] = (f16)(d7 * inv * g1.w + b1.w);
  *(h4*)(o + lane * 4)       = o0;
  *(h4*)(o + 128 + lane * 4) = o1;
}

// ---------------------------------------------------------------- generic WMMA GEMM
// C[M,N] = act(A[M,K] @ W[N,K]^T + bias), block tile 128x64, 8 waves -> 32x32 each.
// Full tiles run an async-LDS double-buffered pipeline (global_load_async_to_lds);
// ragged tiles (readout corners) use the synchronous zero-filled path.
template <bool GELU, bool RESID, bool OUT16, bool SCATTER>
__global__ __launch_bounds__(256) void
k_gemm(const f16* __restrict__ A, int M, int K,
       const f16* __restrict__ W, int N,
       const float* __restrict__ bias,
       f16* __restrict__ out16, int ldo,
       float* __restrict__ resid,
       float* __restrict__ outf,
       const int* __restrict__ gather,
       const int* __restrict__ oidx) {
  __shared__ f16 lA[2][128][48];   // 32 halves + pad, double buffered
  __shared__ f16 lW[2][64][48];

  int tid = threadIdx.x;
  int m0 = blockIdx.y * 128, n0 = blockIdx.x * 64;
  int wid = tid >> 5, lane = tid & 31;
  int wm = (wid >> 1) * 32, wn = (wid & 1) * 32;
  int hf = lane >> 4, ln = lane & 15;
  int aoff = hf ? 8 : 0, koff = hf ? 16 : 0;

  v8f acc00 = {}, acc01 = {}, acc10 = {}, acc11 = {};
  const int KT = K >> 5;

  bool full = (m0 + 128 <= M) && (n0 + 64 <= N);
  if (full) {
    // ---- async double-buffered pipeline
    int rA = tid >> 1, ccA = (tid & 1) * 16;
    int arow = gather ? gather[m0 + rA] : (m0 + rA);
    const f16* gA = A + (long)arow * K + ccA;
    int rW = tid >> 2, ccW = (tid & 3) * 8;
    const f16* gW = W + (long)(n0 + rW) * K + ccW;

    async_load_b128(gA,     &lA[0][rA][ccA]);
    async_load_b128(gA + 8, &lA[0][rA][ccA + 8]);
    async_load_b128(gW,     &lW[0][rW][ccW]);

    for (int t = 0; t < KT; ++t) {
      wait_async0();
      __syncthreads();               // tile t staged; prior compute finished
      if (t + 1 < KT) {
        const f16* pa = gA + (t + 1) * 32;
        const f16* pw = gW + (t + 1) * 32;
        int nb = (t + 1) & 1;
        async_load_b128(pa,     &lA[nb][rA][ccA]);
        async_load_b128(pa + 8, &lA[nb][rA][ccA + 8]);
        async_load_b128(pw,     &lW[nb][rW][ccW]);
      }
      int cb = t & 1;
      v16h a0 = frag_a(&lA[cb][wm + ln][0], aoff);
      v16h a1 = frag_a(&lA[cb][wm + 16 + ln][0], aoff);
      v16h b0 = frag_b(&lW[cb][wn + ln][koff]);
      v16h b1 = frag_b(&lW[cb][wn + 16 + ln][koff]);
      acc00 = wmma16(a0, b0, acc00);
      acc01 = wmma16(a0, b1, acc01);
      acc10 = wmma16(a1, b0, acc10);
      acc11 = wmma16(a1, b1, acc11);
    }
  } else {
    // ---- synchronous path with zero fill for ragged tiles
    for (int t = 0; t < KT; ++t) {
      int k0 = t * 32;
      {
        int r = tid >> 1, cc = (tid & 1) * 16;
        int gr = m0 + r;
        uint4 v0 = {0, 0, 0, 0}, v1 = {0, 0, 0, 0};
        if (gr < M) {
          int ar = gather ? gather[gr] : gr;
          const uint4* src = (const uint4*)(A + (long)ar * K + k0 + cc);
          v0 = src[0];
          v1 = src[1];
        }
        *(uint4*)&lA[0][r][cc]     = v0;
        *(uint4*)&lA[0][r][cc + 8] = v1;
      }
      {
        int r = tid >> 2, cc = (tid & 3) * 8;
        uint4 v = {0, 0, 0, 0};
        if (n0 + r < N) v = *(const uint4*)(W + (long)(n0 + r) * K + k0 + cc);
        *(uint4*)&lW[0][r][cc] = v;
      }
      __syncthreads();
      v16h a0 = frag_a(&lA[0][wm + ln][0], aoff);
      v16h a1 = frag_a(&lA[0][wm + 16 + ln][0], aoff);
      v16h b0 = frag_b(&lW[0][wn + ln][koff]);
      v16h b1 = frag_b(&lW[0][wn + 16 + ln][koff]);
      acc00 = wmma16(a0, b0, acc00);
      acc01 = wmma16(a0, b1, acc01);
      acc10 = wmma16(a1, b0, acc10);
      acc11 = wmma16(a1, b1, acc11);
      __syncthreads();
    }
  }

  v8f* accs[4] = {&acc00, &acc01, &acc10, &acc11};
#pragma unroll
  for (int i = 0; i < 2; ++i)
#pragma unroll
    for (int j = 0; j < 2; ++j) {
      v8f c = *accs[i * 2 + j];
      int nt = n0 + wn + j * 16 + ln;
      if (nt >= N) continue;
      float bz = bias ? bias[nt] : 0.0f;
#pragma unroll
      for (int e = 0; e < 8; ++e) {
        int mt = m0 + wm + i * 16 + hf * 8 + e;
        if (mt >= M) continue;
        float v = c[e] + bz;
        if (GELU) v = 0.5f * v * (1.0f + erff(v * 0.70710678118654752f));
        if (RESID) {
          resid[(long)mt * N + nt] += v;
        } else if (OUT16) {
          out16[(long)mt * ldo + nt] = (f16)v;
        } else if (SCATTER) {
          outf[(long)oidx[mt] + nt] = v;
        }
      }
    }
}

// ---------------------------------------------------------------- fused attention
// grid: (S/16 query tiles, H, B); block 256 threads (8 waves).
__global__ __launch_bounds__(256) void
k_attn(const f16* __restrict__ qkv, const float* __restrict__ mask,
       f16* __restrict__ attn_out) {
  __shared__ f16 kK[SS][40];       // keys, padded
  __shared__ f16 kV[HDIM][264];    // V transposed [dim][key], padded
  __shared__ float sc[16][SS];     // score tile / exp scratch

  int b = blockIdx.z, hh = blockIdx.y, q0 = blockIdx.x * 16;
  int tid = threadIdx.x, wid = tid >> 5, lane = tid & 31;
  int hf = lane >> 4, ln = lane & 15;
  long base = ((long)b * SS) * (3 * DD) + hh * HDIM;

  {  // stage K (async direct-to-LDS) and V (transposed, synchronous)
    int t = tid;
    const f16* kp = qkv + base + (long)t * (3 * DD) + DD;        // K block
    const f16* vp = kp + DD;                                     // V block
    async_load_b128(kp,      &kK[t][0]);
    async_load_b128(kp + 8,  &kK[t][8]);
    async_load_b128(kp + 16, &kK[t][16]);
    async_load_b128(kp + 24, &kK[t][24]);
#pragma unroll
    for (int d2 = 0; d2 < HDIM; ++d2) kV[d2][t] = vp[d2];
  }
  wait_async0();
  __syncthreads();

  // scores: wave handles two 16-key tiles, one 16x16x32 WMMA each
  const f16* qrow = qkv + base + (long)(q0 + ln) * (3 * DD);
  v16h aq = frag_a(qrow, hf ? 8 : 0);
  const float scale = 0.17677669529663687f;  // 1/sqrt(32)
#pragma unroll
  for (int i = 0; i < 2; ++i) {
    int t0 = (wid * 2 + i) * 16;
    v16h bk = frag_b(&kK[t0 + ln][hf ? 16 : 0]);
    v8f c = {};
    c = wmma16(aq, bk, c);
    int tcol = t0 + ln;
#pragma unroll
    for (int e = 0; e < 8; ++e) {
      int m = hf * 8 + e;
      sc[m][tcol] = c[e] * scale + mask[(q0 + m) * SS + tcol];
    }
  }
  __syncthreads();

  // masked softmax: 16 threads per row, 16 cols each; probs overlay sc as f16
  f16* pr = (f16*)&sc[0][0];  // pr[r*264 + c], 4224 halves (8448 B) < 16KB
  {
    int r = tid >> 4, cs = tid & 15;
    float mx = -3.0e38f;
    float ev[16];
#pragma unroll
    for (int u = 0; u < 16; ++u) ev[u] = sc[r][cs + u * 16];
#pragma unroll
    for (int u = 0; u < 16; ++u) mx = fmaxf(mx, ev[u]);
#pragma unroll
    for (int m2 = 8; m2 >= 1; m2 >>= 1) mx = fmaxf(mx, __shfl_xor(mx, m2, 32));
    float sum = 0.0f;
#pragma unroll
    for (int u = 0; u < 16; ++u) { ev[u] = __expf(ev[u] - mx); sum += ev[u]; }
#pragma unroll
    for (int m2 = 8; m2 >= 1; m2 >>= 1) sum += __shfl_xor(sum, m2, 32);
    float inv = 1.0f / sum;
    __syncthreads();  // all sc reads done before pr overlay writes
#pragma unroll
    for (int u = 0; u < 16; ++u) pr[r * 264 + cs + u * 16] = (f16)(ev[u] * inv);
  }
  __syncthreads();

  // A*V: waves 0,1 produce the 16x32 output (one 16-wide n-tile each)
  if (wid < 2) {
    int ntile = wid * 16;
    v8f c = {};
#pragma unroll
    for (int kk = 0; kk < 8; ++kk) {
      v16h ap = frag_a(&pr[ln * 264 + kk * 32], hf ? 8 : 0);
      v16h bv = frag_b(&kV[ntile + ln][kk * 32 + (hf ? 16 : 0)]);
      c = wmma16(ap, bv, c);
    }
    long obase = ((long)b * SS + q0) * DD + hh * HDIM + ntile;
#pragma unroll
    for (int e = 0; e < 8; ++e) {
      int m = hf * 8 + e;
      attn_out[obase + (long)m * DD + ln] = (f16)c[e];
    }
  }
}

// ---------------------------------------------------------------- readout setup
__global__ void k_setup(const int* __restrict__ cidx, const int* __restrict__ eidx,
                        const int* __restrict__ bidx,
                        int* __restrict__ cg, int* __restrict__ co,
                        int* __restrict__ eg, int* __restrict__ eo,
                        int* __restrict__ bg, int* __restrict__ bo,
                        int out_stride) {
  __shared__ int od[SS];
  __shared__ int off[SS];
  int tid = threadIdx.x;
  int v = 0;
  for (int i = 0; i < 4; ++i)   if (cidx[i] == tid) v = 32;
  for (int i = 0; i < 56; ++i)  if (eidx[i] == tid) v = 128;
  for (int i = 0; i < 196; ++i) if (bidx[i] == tid) v = 512;
  od[tid] = v;
  __syncthreads();
  if (tid == 0) {
    int a = 0;
    for (int i = 0; i < SS; ++i) { off[i] = a; a += od[i]; }
  }
  __syncthreads();
  {  // corners: 4 groups x 64 batch = 256 rows
    int g = tid >> 6, b2 = tid & 63;
    int s2 = cidx[g];
    cg[tid] = b2 * SS + s2;
    co[tid] = b2 * out_stride + off[s2];
  }
  for (int r = tid; r < 4 * 14 * BB; r += 256) {  // edges
    int g = r / (14 * BB), rem = r % (14 * BB), b2 = rem / 14, n = rem % 14;
    int s2 = eidx[g * 14 + n];
    eg[r] = b2 * SS + s2;
    eo[r] = b2 * out_stride + off[s2];
  }
  for (int r = tid; r < BB * 196; r += 256) {      // bulk
    int b2 = r / 196, i = r % 196;
    int s2 = bidx[i];
    bg[r] = b2 * SS + s2;
    bo[r] = b2 * out_stride + off[s2];
  }
}

// ---------------------------------------------------------------- host
extern "C" void kernel_launch(void* const* d_in, const int* in_sizes, int n_in,
                              void* d_out, int out_size, void* d_ws, size_t ws_size,
                              hipStream_t stream) {
  const int*   x        = (const int*)d_in[0];
  const float* semb     = (const float*)d_in[1];
  const float* mask     = (const float*)d_in[2];
  const float* ln1_w    = (const float*)d_in[3];
  const float* ln1_b    = (const float*)d_in[4];
  const float* qkv_w    = (const float*)d_in[5];
  const float* qkv_b    = (const float*)d_in[6];
  const float* out_w    = (const float*)d_in[7];
  const float* out_b    = (const float*)d_in[8];
  const float* ln2_w    = (const float*)d_in[9];
  const float* ln2_b    = (const float*)d_in[10];
  const float* f1_w     = (const float*)d_in[11];
  const float* f1_b     = (const float*)d_in[12];
  const float* f2_w     = (const float*)d_in[13];
  const float* f2_b     = (const float*)d_in[14];
  const float* fn_w     = (const float*)d_in[15];
  const float* fn_b     = (const float*)d_in[16];
  const float* corner_w = (const float*)d_in[17];
  const float* corner_b = (const float*)d_in[18];
  const float* edge_w   = (const float*)d_in[19];
  const float* edge_b   = (const float*)d_in[20];
  const float* bulk_w   = (const float*)d_in[21];
  const float* bulk_b   = (const float*)d_in[22];
  const int*   cidx     = (const int*)d_in[23];
  const int*   eidx     = (const int*)d_in[24];
  const int*   bidx     = (const int*)d_in[25];
  float* out = (float*)d_out;
  int out_stride = out_size / BB;

  // ---- workspace carve (all offsets 256B aligned)
  char* wsp = (char*)d_ws;
  auto carve = [&](size_t bytes) -> void* {
    void* p = (void*)wsp;
    wsp += (bytes + 255) & ~(size_t)255;
    return p;
  };
  float* h     = (float*)carve((size_t)NROWS * DD * 4);
  f16* xln     = (f16*)carve((size_t)NROWS * DD * 2);
  f16* qkvb    = (f16*)carve((size_t)NROWS * 3 * DD * 2);
  f16* attnb   = (f16*)carve((size_t)NROWS * DD * 2);
  f16* ffb     = (f16*)carve((size_t)NROWS * FFD * 2);
  f16* w_qkv   = (f16*)carve((size_t)NLAY * 3 * DD * DD * 2);
  f16* w_out   = (f16*)carve((size_t)NLAY * DD * DD * 2);
  f16* w_f1    = (f16*)carve((size_t)NLAY * FFD * DD * 2);
  f16* w_f2    = (f16*)carve((size_t)NLAY * DD * FFD * 2);
  f16* w_cor   = (f16*)carve((size_t)4 * 32 * DD * 2);
  f16* w_edg   = (f16*)carve((size_t)4 * 128 * DD * 2);
  f16* w_blk   = (f16*)carve((size_t)512 * DD * 2);
  int* cg = (int*)carve(256 * 4);
  int* co = (int*)carve(256 * 4);
  int* eg = (int*)carve(3584 * 4);
  int* eo = (int*)carve(3584 * 4);
  int* bg = (int*)carve(12544 * 4);
  int* bo = (int*)carve(12544 * 4);

  auto conv = [&](const float* src, f16* dst, int n) {
    k_f32_to_f16<<<(n + 255) / 256, 256, 0, stream>>>(src, dst, n);
  };
  conv(qkv_w,    w_qkv, NLAY * 3 * DD * DD);
  conv(out_w,    w_out, NLAY * DD * DD);
  conv(f1_w,     w_f1,  NLAY * FFD * DD);
  conv(f2_w,     w_f2,  NLAY * DD * FFD);
  conv(corner_w, w_cor, 4 * 32 * DD);
  conv(edge_w,   w_edg, 4 * 128 * DD);
  conv(bulk_w,   w_blk, 512 * DD);

  k_setup<<<1, 256, 0, stream>>>(cidx, eidx, bidx, cg, co, eg, eo, bg, bo,
                                 out_stride);
  k_embed<<<NROWS, 256, 0, stream>>>(x, semb, h);

  for (int l = 0; l < NLAY; ++l) {
    k_ln<<<NROWS / 8, 256, 0, stream>>>(h, ln1_w + l * DD, ln1_b + l * DD, xln,
                                        NROWS);
    // QKV: (16384 x 256) @ (768 x 256)^T -> f16
    k_gemm<false, false, true, false><<<dim3(12, 128), 256, 0, stream>>>(
        xln, NROWS, DD, w_qkv + (size_t)l * 3 * DD * DD, 3 * DD,
        qkv_b + l * 3 * DD, qkvb, 3 * DD, nullptr, nullptr, nullptr, nullptr);
    // fused attention
    k_attn<<<dim3(SS / 16, HH, BB), 256, 0, stream>>>(qkvb, mask, attnb);
    // out-proj + residual into f32 h
    k_gemm<false, true, false, false><<<dim3(4, 128), 256, 0, stream>>>(
        attnb, NROWS, DD, w_out + (size_t)l * DD * DD, DD, out_b + l * DD,
        nullptr, 0, h, nullptr, nullptr, nullptr);
    k_ln<<<NROWS / 8, 256, 0, stream>>>(h, ln2_w + l * DD, ln2_b + l * DD, xln,
                                        NROWS);
    // FFN up + exact GELU -> f16
    k_gemm<true, false, true, false><<<dim3(16, 128), 256, 0, stream>>>(
        xln, NROWS, DD, w_f1 + (size_t)l * FFD * DD, FFD, f1_b + l * FFD, ffb,
        FFD, nullptr, nullptr, nullptr, nullptr);
    // FFN down + residual
    k_gemm<false, true, false, false><<<dim3(4, 128), 256, 0, stream>>>(
        ffb, NROWS, FFD, w_f2 + (size_t)l * DD * FFD, DD, f2_b + l * DD,
        nullptr, 0, h, nullptr, nullptr, nullptr);
  }

  // final LN
  k_ln<<<NROWS / 8, 256, 0, stream>>>(h, fn_w, fn_b, xln, NROWS);

  // readout heads: gather rows of xln, scatter f32 into d_out
  for (int g = 0; g < 4; ++g) {  // corners: M=64, N=32 (ragged -> sync path)
    k_gemm<false, false, false, true><<<dim3(1, 1), 256, 0, stream>>>(
        xln, BB, DD, w_cor + (size_t)g * 32 * DD, 32, corner_b + g * 32,
        nullptr, 0, nullptr, out, cg + g * BB, co + g * BB);
  }
  for (int g = 0; g < 4; ++g) {  // edges: M=896, N=128 (full tiles -> async)
    k_gemm<false, false, false, true><<<dim3(2, 7), 256, 0, stream>>>(
        xln, 14 * BB, DD, w_edg + (size_t)g * 128 * DD, 128, edge_b + g * 128,
        nullptr, 0, nullptr, out, eg + g * 14 * BB, eo + g * 14 * BB);
  }
  // bulk: M=12544, N=512
  k_gemm<false, false, false, true><<<dim3(8, 98), 256, 0, stream>>>(
      xln, BB * 196, DD, w_blk, 512, bulk_b, nullptr, 0, nullptr, out, bg, bo);
}
